// MultiHeadAttention_41669772706241
// MI455X (gfx1250) — matrix operand order repaired and checked by reading, not compiled, not run on hardware
//
#include <hip/hip_runtime.h>
#include <math.h>

// ---------------------------------------------------------------------------
// MI455X (gfx1250) multi-head attention forward.
// - All matmuls on v_wmma_f32_16x16x32_bf16 (wave32, f32 accumulate).
// - attn_weights (537MB f32) is a mandatory output -> memory-bound on that
//   stream; the PV GEMM re-reads it through the Tensor Data Mover into LDS
//   (double-buffered, padded rows for bank-conflict-free fragment gathers).
// ---------------------------------------------------------------------------

typedef __attribute__((ext_vector_type(16))) __bf16        v16bf;
typedef __attribute__((ext_vector_type(8)))  float         v8f;
typedef __attribute__((ext_vector_type(4)))  unsigned int  v4u;
typedef __attribute__((ext_vector_type(4)))  float         v4f;
typedef __attribute__((ext_vector_type(8)))  int           i32x8;
typedef __attribute__((ext_vector_type(4)))  int           i32x4;

union FragU { v16bf f; v4u u4[2]; unsigned short s[16]; };

__device__ __forceinline__ unsigned short f2bf(float x) {
  union { float f; unsigned int u; } c; c.f = x;
  unsigned int u = c.u;
  return (unsigned short)((u + 0x7FFFu + ((u >> 16) & 1u)) >> 16); // RNE
}

__device__ __forceinline__ v8f wmma_bf16(v16bf a, v16bf b, v8f c) {
  return __builtin_amdgcn_wmma_f32_16x16x32_bf16(
      /*neg_a=*/false, a, /*neg_b=*/false, b,
      /*c_mod=*/(short)0, c, /*reuse_a=*/false, /*reuse_b=*/false);
}

// A fragment: 16x32 bf16 tile, row-major, lane&15 = M.
// K chunks: [half*8, half*8+8) and [16+half*8, 16+half*8+8).
__device__ __forceinline__ v16bf load_a_bf16(const unsigned short* tile, int ld, int lane) {
  const int r = lane & 15, half = lane >> 4;
  const unsigned short* p = tile + (size_t)r * ld + half * 8;
  FragU fu;
  fu.u4[0] = *(const v4u*)p;
  fu.u4[1] = *(const v4u*)(p + 16);
  return fu.f;
}

// A fragment from an f32 tile in LDS (ld in elements), converted to bf16.
__device__ __forceinline__ v16bf load_a_f32_lds(const float* tile, int ld, int lane) {
  const int r = lane & 15, half = lane >> 4;
  const float* p = tile + r * ld + half * 8;
  FragU fu;
#pragma unroll
  for (int i = 0; i < 8; ++i) fu.s[i]     = f2bf(p[i]);
#pragma unroll
  for (int i = 0; i < 8; ++i) fu.s[8 + i] = f2bf(p[16 + i]);
  return fu.f;
}

// B fragment: B is Kx16; source is B^T row-major ([N,K]) so lane&15 = N reads
// K chunks [half*8, half*8+8) and [16+half*8, 16+half*8+8).
__device__ __forceinline__ v16bf load_b_bf16(const unsigned short* tile, int ld, int lane) {
  const int n = lane & 15, half = lane >> 4;
  const unsigned short* p = tile + (size_t)n * ld + half * 8;
  FragU fu;
  fu.u4[0] = *(const v4u*)p;
  fu.u4[1] = *(const v4u*)(p + 16);
  return fu.f;
}

// --------------------- Tensor Data Mover 2D f32 tile load ------------------
// Builds a D# (groups 0/1 per CDNA5 ISA) for a 2D f32 tile:
//   tile_w x tile_h elements, row stride stride_elems, LDS row padding of
//   1 DWORD every 32 DWORDs (pad_interval code 4, pad_amount code 0).
// Issued by one wave; tracked with TENSORcnt.
// This toolchain's builtin takes 6 args (g0, g1, g2, g3, extra, cpol);
// groups 2/3 and the extra words are zero for a <=2D tensor.
__device__ __forceinline__ void tdm_load_tile_f32(
    unsigned lds_byte_addr, const float* gptr,
    unsigned tile_w, unsigned tile_h,
    unsigned tensor_w, unsigned tensor_h, unsigned stride_elems) {
  const unsigned long long ga = (unsigned long long)(size_t)gptr;
  v4u g0;
  g0[0] = 1u;                                            // count=1, user mode
  g0[1] = lds_byte_addr;                                 // lds_addr
  g0[2] = (unsigned)(ga & 0xFFFFFFFFu);                  // global_addr[31:0]
  g0[3] = (unsigned)((ga >> 32) & 0x01FFFFFFu) | (2u << 30); // addr[56:32], type=2
  i32x8 g1;
  g1[0] = (int)((2u << 16) | (1u << 20) | (4u << 22));   // 4B elems, pad_en, every 32 DW, +1 DW
  g1[1] = (int)((tensor_w & 0xFFFFu) << 16);             // tensor_dim0[15:0]
  g1[2] = (int)(((tensor_w >> 16) & 0xFFFFu) | ((tensor_h & 0xFFFFu) << 16));
  g1[3] = (int)(((tensor_h >> 16) & 0xFFFFu) | ((tile_w & 0xFFFFu) << 16));
  g1[4] = (int)(tile_h & 0xFFFFu);                       // tile_dim1 (tile_dim2=0)
  g1[5] = (int)stride_elems;                             // tensor_dim0_stride[31:0]
  g1[6] = 0;
  g1[7] = 0;
  const i32x4 z4 = {0, 0, 0, 0};                         // groups 2/3 unused (<=2D)
  const i32x8 z8 = {0, 0, 0, 0, 0, 0, 0, 0};
  __builtin_amdgcn_tensor_load_to_lds(g0, g1, z4, z4, z8, 0);
}

// --------------------------- f32 -> bf16 convert ---------------------------
__global__ void cvt_f32_bf16_kernel(const float* __restrict__ in,
                                    unsigned short* __restrict__ out, int n) {
  int i = blockIdx.x * blockDim.x + threadIdx.x;
  int stride = gridDim.x * blockDim.x;
  for (; i < n; i += stride) out[i] = f2bf(in[i]);
}

// --------------------------- generic y = A @ W^T ---------------------------
// A: [M,K] bf16 row-major.  Wt: [N,K] bf16 row-major (nn.Linear weight).
// Block: 256 thr (8 waves), tile 128x128, wave tile 32x64 (2x4 WMMA tiles).
// MODE 0: f32 out row-major [M,N]
// MODE 1: bf16 out scattered to [b,h,t,dk]   (Q, K)
// MODE 2: bf16 out scattered to [b,h,dk,t]   (V transposed)
template <int MODE>
__global__ __launch_bounds__(256) void gemm_xw_kernel(
    const unsigned short* __restrict__ A, const unsigned short* __restrict__ Wt,
    void* __restrict__ out, int M, int N, int K) {
  const int tid = threadIdx.x, lane = tid & 31, wid = tid >> 5;
  const int waveM = wid >> 1, waveN = wid & 1;
  const int wm = blockIdx.y * 128 + waveM * 32;
  const int wn = blockIdx.x * 128 + waveN * 64;

  v8f acc[2][4];
#pragma unroll
  for (int i = 0; i < 2; ++i)
#pragma unroll
    for (int j = 0; j < 4; ++j) acc[i][j] = v8f{0.f,0.f,0.f,0.f,0.f,0.f,0.f,0.f};

  for (int k0 = 0; k0 < K; k0 += 32) {
    __builtin_prefetch(A + (size_t)wm * K + k0 + 32, 0, 1);
    v16bf a[2], b[4];
#pragma unroll
    for (int i = 0; i < 2; ++i)
      a[i] = load_a_bf16(A + (size_t)(wm + 16 * i) * K + k0, K, lane);
#pragma unroll
    for (int j = 0; j < 4; ++j)
      b[j] = load_b_bf16(Wt + (size_t)(wn + 16 * j) * K + k0, K, lane);
#pragma unroll
    for (int i = 0; i < 2; ++i)
#pragma unroll
      for (int j = 0; j < 4; ++j) acc[i][j] = wmma_bf16(a[i], b[j], acc[i][j]);
  }

  const int half = lane >> 4, nl = lane & 15;
#pragma unroll
  for (int i = 0; i < 2; ++i)
#pragma unroll
    for (int j = 0; j < 4; ++j)
#pragma unroll
      for (int r = 0; r < 8; ++r) {
        const int m = wm + 16 * i + r + 8 * half;
        const int n = wn + 16 * j + nl;
        const float v = acc[i][j][r];
        if (MODE == 0) {
          ((float*)out)[(size_t)m * N + n] = v;
        } else if (MODE == 1) {  // [b,h,t,dk], T=2048, dk=64, H=16
          const size_t idx = (((size_t)(m >> 11) * 16 + (n >> 6)) * 2048 + (m & 2047)) * 64 + (n & 63);
          ((unsigned short*)out)[idx] = f2bf(v);
        } else {                 // [b,h,dk,t]
          const size_t idx = (((size_t)(m >> 11) * 16 + (n >> 6)) * 64 + (n & 63)) * 2048 + (m & 2047);
          ((unsigned short*)out)[idx] = f2bf(v);
        }
      }
}

// --------------------------- S = Q @ K^T * 1/8 -----------------------------
// Q,K: [32][2048][64] bf16.  attn out: [32][2048][2048] f32.
__global__ __launch_bounds__(256) void scores_kernel(
    const unsigned short* __restrict__ Q, const unsigned short* __restrict__ Kh,
    float* __restrict__ attn) {
  const int tid = threadIdx.x, lane = tid & 31, wid = tid >> 5;
  const int waveM = wid >> 1, waveN = wid & 1;
  const int bh = blockIdx.z;
  const unsigned short* Qb = Q  + (size_t)bh * 2048 * 64;
  const unsigned short* Kb = Kh + (size_t)bh * 2048 * 64;
  const int qm = blockIdx.y * 128 + waveM * 32;
  const int kn = blockIdx.x * 128 + waveN * 64;

  v8f acc[2][4];
#pragma unroll
  for (int i = 0; i < 2; ++i)
#pragma unroll
    for (int j = 0; j < 4; ++j) acc[i][j] = v8f{0.f,0.f,0.f,0.f,0.f,0.f,0.f,0.f};

#pragma unroll
  for (int k0 = 0; k0 < 64; k0 += 32) {
    v16bf a[2], b[4];
#pragma unroll
    for (int i = 0; i < 2; ++i)
      a[i] = load_a_bf16(Qb + (size_t)(qm + 16 * i) * 64 + k0, 64, lane);
#pragma unroll
    for (int j = 0; j < 4; ++j)
      b[j] = load_b_bf16(Kb + (size_t)(kn + 16 * j) * 64 + k0, 64, lane);
#pragma unroll
    for (int i = 0; i < 2; ++i)
#pragma unroll
      for (int j = 0; j < 4; ++j) acc[i][j] = wmma_bf16(a[i], b[j], acc[i][j]);
  }

  const int half = lane >> 4, nl = lane & 15;
  float* Ab = attn + (size_t)bh * 2048 * 2048;
#pragma unroll
  for (int i = 0; i < 2; ++i)
#pragma unroll
    for (int j = 0; j < 4; ++j)
#pragma unroll
      for (int r = 0; r < 8; ++r) {
        const int m = qm + 16 * i + r + 8 * half;
        const int n = kn + 16 * j + nl;
        Ab[(size_t)m * 2048 + n] = 0.125f * acc[i][j][r];  // 1/sqrt(64)
      }
}

// --------------------------- row softmax (len 2048) ------------------------
__global__ __launch_bounds__(256) void softmax_kernel(float* __restrict__ attn) {
  float* p = attn + (size_t)blockIdx.x * 2048;
  const int tid = threadIdx.x;
  __shared__ float red[256];

  float v[8];
  v4f x0 = *(const v4f*)(p + tid * 8);
  v4f x1 = *(const v4f*)(p + tid * 8 + 4);
#pragma unroll
  for (int i = 0; i < 4; ++i) { v[i] = x0[i]; v[4 + i] = x1[i]; }

  float mx = v[0];
#pragma unroll
  for (int i = 1; i < 8; ++i) mx = fmaxf(mx, v[i]);
  red[tid] = mx;
  __syncthreads();
  for (int s = 128; s > 0; s >>= 1) {
    if (tid < s) red[tid] = fmaxf(red[tid], red[tid + s]);
    __syncthreads();
  }
  const float rmax = red[0];
  __syncthreads();

  float sum = 0.f;
#pragma unroll
  for (int i = 0; i < 8; ++i) { v[i] = __expf(v[i] - rmax); sum += v[i]; }
  red[tid] = sum;
  __syncthreads();
  for (int s = 128; s > 0; s >>= 1) {
    if (tid < s) red[tid] += red[tid + s];
    __syncthreads();
  }
  const float inv = 1.f / red[0];

  v4f y0, y1;
#pragma unroll
  for (int i = 0; i < 4; ++i) { y0[i] = v[i] * inv; y1[i] = v[4 + i] * inv; }
  *(v4f*)(p + tid * 8)     = y0;
  *(v4f*)(p + tid * 8 + 4) = y1;
}

// --------------------------- O = attn @ V ----------------------------------
// attn: [32][2048][2048] f32. A-tiles (128x32 f32) are DMA'd by the Tensor
// Data Mover into LDS (double-buffered, row padded 32->33 DWORDs so the
// 16-lane fragment gather is bank-conflict-free), then converted to bf16.
// Vt: [32][64][2048] bf16 (B^T row-major -> contiguous B-frags from global).
// ctx: [B*T, 1024] bf16, head h writes columns [h*64, h*64+64).
#define PV_LDS_LD 33  // padded row stride in floats
__global__ __launch_bounds__(256) void pv_kernel(
    const float* __restrict__ attn, const unsigned short* __restrict__ Vt,
    unsigned short* __restrict__ ctx) {
  __shared__ float tile[2][128 * PV_LDS_LD];

  const int tid = threadIdx.x, lane = tid & 31, wid = tid >> 5;
  const int waveM = wid >> 1, waveN = wid & 1;  // 4x2 waves -> 128x64 block tile
  const int bh = blockIdx.z, b = bh >> 4, h = bh & 15;
  const int qm0 = blockIdx.x * 128;
  const float* Ab = attn + (size_t)bh * 2048 * 2048 + (size_t)qm0 * 2048;
  const unsigned short* Vb = Vt + (size_t)bh * 64 * 2048;
  const int dn = waveN * 32;

  v8f acc[2][2];
#pragma unroll
  for (int i = 0; i < 2; ++i)
#pragma unroll
    for (int j = 0; j < 2; ++j) acc[i][j] = v8f{0.f,0.f,0.f,0.f,0.f,0.f,0.f,0.f};

  // Prologue: wave 0 DMAs the first 128x32 tile, waits, whole WG syncs.
  if (wid == 0) {
    tdm_load_tile_f32((unsigned)(size_t)&tile[0][0], Ab, 32, 128, 2048, 2048, 2048);
    __builtin_amdgcn_s_wait_tensorcnt(0);
  }
  __syncthreads();

  int cur = 0;
  for (int k0 = 0; k0 < 2048; k0 += 32) {
    // Kick off DMA of the next tile into the other buffer.
    if (k0 + 32 < 2048 && wid == 0)
      tdm_load_tile_f32((unsigned)(size_t)&tile[cur ^ 1][0], Ab + k0 + 32,
                        32, 128, 2048, 2048, 2048);

    v16bf a[2], bfr[2];
#pragma unroll
    for (int i = 0; i < 2; ++i)
      a[i] = load_a_f32_lds(&tile[cur][(waveM * 32 + 16 * i) * PV_LDS_LD], PV_LDS_LD, lane);
#pragma unroll
    for (int j = 0; j < 2; ++j)
      bfr[j] = load_b_bf16(Vb + (size_t)(dn + 16 * j) * 2048 + k0, 2048, lane);
#pragma unroll
    for (int i = 0; i < 2; ++i)
#pragma unroll
      for (int j = 0; j < 2; ++j) acc[i][j] = wmma_bf16(a[i], bfr[j], acc[i][j]);

    if (wid == 0) __builtin_amdgcn_s_wait_tensorcnt(0);
    __syncthreads();
    cur ^= 1;
  }

  const int half = lane >> 4, nl = lane & 15;
#pragma unroll
  for (int i = 0; i < 2; ++i)
#pragma unroll
    for (int j = 0; j < 2; ++j)
#pragma unroll
      for (int r = 0; r < 8; ++r) {
        const int q  = qm0 + waveM * 32 + 16 * i + r + 8 * half;
        const int dd = dn + 16 * j + nl;
        ctx[((size_t)b * 2048 + q) * 1024 + h * 64 + dd] = f2bf(acc[i][j][r]);
      }
}

// ---------------------------------------------------------------------------
extern "C" void kernel_launch(void* const* d_in, const int* in_sizes, int n_in,
                              void* d_out, int out_size, void* d_ws, size_t ws_size,
                              hipStream_t stream) {
  const float* x  = (const float*)d_in[0];  // [2,2048,1024]
  const float* Wq = (const float*)d_in[1];  // [1024,1024]
  const float* Wk = (const float*)d_in[2];
  const float* Wv = (const float*)d_in[3];
  const float* Wo = (const float*)d_in[4];

  float* out  = (float*)d_out;                       // [2,2048,1024]
  float* attn = out + (size_t)2 * 2048 * 1024;       // [2,16,2048,2048]

  // Workspace layout (bf16 as raw ushort), ~48 MB total.
  unsigned short* ws  = (unsigned short*)d_ws;
  unsigned short* xb  = ws;                          // 4096*1024
  unsigned short* wqb = xb  + (size_t)4096 * 1024;
  unsigned short* wkb = wqb + (size_t)1024 * 1024;
  unsigned short* wvb = wkb + (size_t)1024 * 1024;
  unsigned short* wob = wvb + (size_t)1024 * 1024;
  unsigned short* Qb  = wob + (size_t)1024 * 1024;   // [32][2048][64]
  unsigned short* Kb  = Qb  + (size_t)32 * 2048 * 64;
  unsigned short* Vtb = Kb  + (size_t)32 * 2048 * 64; // [32][64][2048]
  unsigned short* ctx = Vtb + (size_t)32 * 2048 * 64; // [4096][1024]

  cvt_f32_bf16_kernel<<<2048, 256, 0, stream>>>(x,  xb,  4096 * 1024);
  cvt_f32_bf16_kernel<<<1024, 256, 0, stream>>>(Wq, wqb, 1024 * 1024);
  cvt_f32_bf16_kernel<<<1024, 256, 0, stream>>>(Wk, wkb, 1024 * 1024);
  cvt_f32_bf16_kernel<<<1024, 256, 0, stream>>>(Wv, wvb, 1024 * 1024);
  cvt_f32_bf16_kernel<<<1024, 256, 0, stream>>>(Wo, wob, 1024 * 1024);

  const dim3 gProj(1024 / 128, 4096 / 128);  // (N tiles, M tiles)
  gemm_xw_kernel<1><<<gProj, 256, 0, stream>>>(xb, wqb, Qb,  4096, 1024, 1024);
  gemm_xw_kernel<1><<<gProj, 256, 0, stream>>>(xb, wkb, Kb,  4096, 1024, 1024);
  gemm_xw_kernel<2><<<gProj, 256, 0, stream>>>(xb, wvb, Vtb, 4096, 1024, 1024);

  scores_kernel<<<dim3(16, 16, 32), 256, 0, stream>>>(Qb, Kb, attn);
  softmax_kernel<<<32 * 2048, 256, 0, stream>>>(attn);
  pv_kernel<<<dim3(16, 1, 32), 256, 0, stream>>>(attn, Vtb, ctx);

  gemm_xw_kernel<0><<<gProj, 256, 0, stream>>>(ctx, wob, out, 4096, 1024, 1024);
}
